// AlgorithmicNoiseLayer_81389630259917
// MI455X (gfx1250) — matrix operand order repaired and checked by reading, not compiled, not run on hardware
//
#include <hip/hip_runtime.h>

// ---------------------------------------------------------------------------
// Types for CDNA5 WMMA
// ---------------------------------------------------------------------------
typedef __attribute__((ext_vector_type(16))) __bf16 v16bf;
typedef __attribute__((ext_vector_type(8)))  float  v8f;

__device__ __forceinline__ unsigned short f2bf(float f) {
    unsigned int u = __float_as_uint(f);
    u += 0x7FFFu + ((u >> 16) & 1u);       // round-to-nearest-even
    return (unsigned short)(u >> 16);
}

// ---------------------------------------------------------------------------
// CDNA5 async global->LDS (ASYNCcnt-tracked DMA, no VGPR staging)
//   global_load_async_to_lds_b128 vdst(LDS byte addr), vaddr(64b), off
// ---------------------------------------------------------------------------
__device__ __forceinline__ void async_copy_b128(unsigned lds_byte_addr,
                                                const void* gptr) {
    asm volatile("global_load_async_to_lds_b128 %0, %1, off"
                 :
                 : "v"(lds_byte_addr),
                   "v"((unsigned long long)(uintptr_t)gptr)
                 : "memory");
}
__device__ __forceinline__ void wait_async0() {
    asm volatile("s_wait_asynccnt 0x0" ::: "memory");
}
// low 32 bits of a generic LDS-aperture address == wave-relative LDS offset
__device__ __forceinline__ unsigned lds_addr(const void* p) {
    return (unsigned)(uintptr_t)p;
}

// ---------------------------------------------------------------------------
// 1) Deterministic pool-random noise (sequential LCG; 1 thread)
// ---------------------------------------------------------------------------
__global__ void noise_kernel(float* __restrict__ noise, int n) {
    __shared__ unsigned int pool[4096];
    if (threadIdx.x != 0 || blockIdx.x != 0 || n > 4096) return;
    unsigned int s = 123u;
    for (int i = 0; i < n; ++i) { s = 65539u * s + 1u; pool[i] = s; }
    int nxt = n - 1;
    for (int i = 0; i < n; ++i) {
        nxt = (int)(pool[nxt] % (unsigned)n);
        noise[i] = (float)((double)pool[nxt] * 1e-9);
        s = 65539u * s + 1u;
        pool[nxt] = s;
    }
}

__global__ void zero_kernel(float* __restrict__ p, int n) {
    int i = blockIdx.x * blockDim.x + threadIdx.x;
    if (i < n) p[i] = 0.f;
}

// ---------------------------------------------------------------------------
// 2) Per-channel batch statistics of h = relu(x + noise)
// ---------------------------------------------------------------------------
__global__ __launch_bounds__(256) void stats_kernel(
        const float* __restrict__ x, const float* __restrict__ noise,
        float* __restrict__ sum, float* __restrict__ sumsq,
        int C, int rowsPerBlock) {
    int c = blockIdx.x * blockDim.x + threadIdx.x;
    float nz = noise[c];
    float s1 = 0.f, s2 = 0.f;
    int r0 = blockIdx.y * rowsPerBlock;
    for (int r = 0; r < rowsPerBlock; ++r) {
        float h = fmaxf(x[(size_t)(r0 + r) * C + c] + nz, 0.f);
        s1 += h; s2 += h * h;
    }
    atomicAdd(&sum[c], s1);
    atomicAdd(&sumsq[c], s2);
}

__global__ void finalize_kernel(const float* __restrict__ sum,
                                const float* __restrict__ sumsq,
                                const float* __restrict__ gamma,
                                const float* __restrict__ beta,
                                float* __restrict__ ascale,
                                float* __restrict__ dshift, int C, int B) {
    int c = blockIdx.x * blockDim.x + threadIdx.x;
    if (c >= C) return;
    float invB  = 1.f / (float)B;
    float mean  = sum[c] * invB;
    float var   = sumsq[c] * invB - mean * mean;
    float a     = gamma[c] * rsqrtf(var + 0.8f);
    ascale[c]   = a;
    dshift[c]   = beta[c] - a * mean;
}

// ---------------------------------------------------------------------------
// 3) h_bf16 = bf16(relu(x + noise))   (float4 in, 4x bf16 out)
// ---------------------------------------------------------------------------
__global__ __launch_bounds__(256) void hconv_kernel(
        const float* __restrict__ x, const float* __restrict__ noise,
        unsigned short* __restrict__ hbf, int C, long long n4) {
    long long i = (long long)blockIdx.x * blockDim.x + threadIdx.x;
    if (i >= n4) return;
    float4 v = ((const float4*)x)[i];
    int c = (int)((i * 4) % C);
    float h0 = fmaxf(v.x + noise[c + 0], 0.f);
    float h1 = fmaxf(v.y + noise[c + 1], 0.f);
    float h2 = fmaxf(v.z + noise[c + 2], 0.f);
    float h3 = fmaxf(v.w + noise[c + 3], 0.f);
    uint2 r;
    r.x = (unsigned)f2bf(h0) | ((unsigned)f2bf(h1) << 16);
    r.y = (unsigned)f2bf(h2) | ((unsigned)f2bf(h3) << 16);
    ((uint2*)hbf)[i] = r;
}

// ---------------------------------------------------------------------------
// 4) W'[o,c] = bf16(W[o,c]*a[c]) ;  bias'[o] = b[o] + sum_c d[c]*W[o,c]
// ---------------------------------------------------------------------------
__global__ __launch_bounds__(256) void wprep_kernel(
        const float* __restrict__ W, const float* __restrict__ ascale,
        const float* __restrict__ dshift, const float* __restrict__ b,
        unsigned short* __restrict__ wp, float* __restrict__ biasP, int C) {
    int o = blockIdx.x, tid = threadIdx.x;
    __shared__ float red[256];
    float acc = 0.f;
    for (int c = tid; c < C; c += 256) {
        float w = W[(size_t)o * C + c];
        wp[(size_t)o * C + c] = f2bf(w * ascale[c]);
        acc += w * dshift[c];
    }
    red[tid] = acc;
    __syncthreads();
    for (int s = 128; s > 0; s >>= 1) {
        if (tid < s) red[tid] += red[tid + s];
        __syncthreads();
    }
    if (tid == 0) biasP[o] = b[o] + red[0];
}

// ---------------------------------------------------------------------------
// 5) GEMM: out[M,N] = hbf[M,K](bf16) @ wp[N,K](bf16)^T + bias'  (f32 accum)
//    128x128 block tile, 8 waves (2x4), 64x32 per wave, K step 32.
//    Double-buffered LDS filled by global_load_async_to_lds_b128 (ASYNCcnt),
//    core = v_wmma_f32_16x16x32_bf16.
// ---------------------------------------------------------------------------
#define BM 128
#define BN 128
#define BK 32
#define LK 40   // padded LDS K-stride (elements) to stagger banks

__global__ __launch_bounds__(256) void gemm_kernel(
        const unsigned short* __restrict__ A,   // [M,K] bf16 bits
        const unsigned short* __restrict__ Bm,  // [N,K] bf16 bits
        const float* __restrict__ biasP,
        float* __restrict__ out, int M, int K, int N) {
    __shared__ unsigned short As[2][BM * LK];
    __shared__ unsigned short Bs[2][BN * LK];

    const int tid  = threadIdx.x;
    const int lane = tid & 31;
    const int w    = tid >> 5;
    const int wm   = w & 1;        // 2 waves along M
    const int wn   = w >> 1;       // 4 waves along N
    const int half = lane >> 4;
    const int l15  = lane & 15;
    const int m0 = blockIdx.y * BM;
    const int n0 = blockIdx.x * BN;

    // ---- accumulators, initialized with folded bias (per-lane column) ----
    v8f acc[4][2];
#pragma unroll
    for (int j = 0; j < 2; ++j) {
        float bv = biasP[n0 + wn * 32 + j * 16 + l15];
        v8f c;
#pragma unroll
        for (int v = 0; v < 8; ++v) c[v] = bv;
#pragma unroll
        for (int i = 0; i < 4; ++i) acc[i][j] = c;
    }

    // ---- per-thread DMA coords: two 16B chunks each for A and B tiles ----
    const int c0 = tid, c1 = tid + 256;
    const int ar0 = c0 >> 2, ao0 = (c0 & 3) * 8;   // row / element offset
    const int ar1 = c1 >> 2, ao1 = (c1 & 3) * 8;
    const unsigned short* Abase = A  + (size_t)m0 * K;
    const unsigned short* Bbase = Bm + (size_t)n0 * K;

    const unsigned ldsA = lds_addr(&As[0][0]);
    const unsigned ldsB = lds_addr(&Bs[0][0]);
    const unsigned aoff0 = (unsigned)(ar0 * LK + ao0) * 2u;  // byte offsets
    const unsigned aoff1 = (unsigned)(ar1 * LK + ao1) * 2u;
    const unsigned bufBytes = BM * LK * 2u;                  // per buffer

    // ---- preload tile 0 into buffer 0 via async DMA ----
    async_copy_b128(ldsA + aoff0, Abase + (size_t)ar0 * K + ao0);
    async_copy_b128(ldsA + aoff1, Abase + (size_t)ar1 * K + ao1);
    async_copy_b128(ldsB + aoff0, Bbase + (size_t)ar0 * K + ao0);
    async_copy_b128(ldsB + aoff1, Bbase + (size_t)ar1 * K + ao1);
    wait_async0();
    __syncthreads();

    const int NK = K / BK;
    for (int kt = 0; kt < NK; ++kt) {
        const int cur = kt & 1;
        const int nxt = cur ^ 1;
        const int k1  = (kt + 1) * BK;

        // ---- fire DMA for the next tile; overlaps with WMMA below ----
        if (kt + 1 < NK) {
            const unsigned nb = (unsigned)nxt * bufBytes;
            async_copy_b128(ldsA + nb + aoff0, Abase + (size_t)ar0 * K + k1 + ao0);
            async_copy_b128(ldsA + nb + aoff1, Abase + (size_t)ar1 * K + k1 + ao1);
            async_copy_b128(ldsB + nb + aoff0, Bbase + (size_t)ar0 * K + k1 + ao0);
            async_copy_b128(ldsB + nb + aoff1, Bbase + (size_t)ar1 * K + k1 + ao1);
            if (kt + 2 < NK) {  // L2 prefetch of the tile after next
                __builtin_prefetch(Abase + (size_t)ar0 * K + k1 + BK + ao0, 0, 1);
                __builtin_prefetch(Bbase + (size_t)ar0 * K + k1 + BK + ao0, 0, 1);
            }
        }

        // ---- fragment loads from LDS (ISA VGPR layouts, 16B each) ----
        v16bf af[4], bfm[2];
#pragma unroll
        for (int i = 0; i < 4; ++i) {
            const unsigned short* p =
                &As[cur][(wm * 64 + i * 16 + l15) * LK + half * 8];
            ((uint4*)&af[i])[0] = *(const uint4*)(p);        // K = h*8 .. h*8+7
            ((uint4*)&af[i])[1] = *(const uint4*)(p + 16);   // K = 16+h*8 ..
        }
#pragma unroll
        for (int j = 0; j < 2; ++j) {
            const unsigned short* p =
                &Bs[cur][(wn * 32 + j * 16 + l15) * LK + half * 8];
            ((uint4*)&bfm[j])[0] = *(const uint4*)(p);       // K = h*8 .. h*8+7
            ((uint4*)&bfm[j])[1] = *(const uint4*)(p + 16);  // K = 16+h*8 ..
        }

        // ---- 8x v_wmma_f32_16x16x32_bf16 ----
#pragma unroll
        for (int i = 0; i < 4; ++i)
#pragma unroll
            for (int j = 0; j < 2; ++j)
                acc[i][j] = __builtin_amdgcn_wmma_f32_16x16x32_bf16(
                    false, af[i], false, bfm[j], (short)0, acc[i][j],
                    false, false);

        // own DMA done, then workgroup-wide visibility
        wait_async0();
        __syncthreads();
    }

    // ---- epilogue: C/D layout -> row = v + 8*half, col = l15 ----
#pragma unroll
    for (int i = 0; i < 4; ++i)
#pragma unroll
        for (int j = 0; j < 2; ++j)
#pragma unroll
            for (int v = 0; v < 8; ++v) {
                int row = m0 + wm * 64 + i * 16 + v + half * 8;
                int col = n0 + wn * 32 + j * 16 + l15;
                out[(size_t)row * N + col] = acc[i][j][v];
            }
}

// ---------------------------------------------------------------------------
// Launch
// ---------------------------------------------------------------------------
extern "C" void kernel_launch(void* const* d_in, const int* in_sizes, int n_in,
                              void* d_out, int out_size, void* d_ws, size_t ws_size,
                              hipStream_t stream) {
    const float* x     = (const float*)d_in[0];
    const float* gamma = (const float*)d_in[1];
    const float* beta  = (const float*)d_in[2];
    const float* W     = (const float*)d_in[3];
    const float* bias  = (const float*)d_in[4];
    float* out = (float*)d_out;

    const int C = in_sizes[1];            // 4096
    const int B = in_sizes[0] / C;        // 8192
    const int O = in_sizes[4];            // 4096

    // workspace layout
    float* wsf    = (float*)d_ws;
    float* noise  = wsf;
    float* sum    = wsf + C;
    float* sumsq  = wsf + 2 * C;
    float* ascale = wsf + 3 * C;
    float* dshift = wsf + 4 * C;
    float* biasP  = wsf + 5 * C;
    unsigned short* hbf = (unsigned short*)(biasP + O);
    unsigned short* wp  = hbf + (size_t)B * C;

    noise_kernel<<<1, 32, 0, stream>>>(noise, C);
    zero_kernel<<<(2 * C + 255) / 256, 256, 0, stream>>>(sum, 2 * C);

    dim3 sg(C / 256, B / 256);
    stats_kernel<<<sg, 256, 0, stream>>>(x, noise, sum, sumsq, C, 256);
    finalize_kernel<<<(C + 255) / 256, 256, 0, stream>>>(sum, sumsq, gamma, beta,
                                                         ascale, dshift, C, B);

    long long n4 = (long long)B * C / 4;
    hconv_kernel<<<(int)((n4 + 255) / 256), 256, 0, stream>>>(x, noise, hbf, C, n4);
    wprep_kernel<<<O, 256, 0, stream>>>(W, ascale, dshift, bias, wp, biasP, C);

    dim3 gg(O / BN, B / BM);              // (32, 64) tiles
    gemm_kernel<<<gg, 256, 0, stream>>>(hbf, wp, biasP, out, B, C, O);
}
